// BiLSTM_83760452207019
// MI455X (gfx1250) — compile-verified, matrix operand order checked by reading, not run on hardware
//
#include <hip/hip_runtime.h>
#include <hip/hip_bf16.h>

#define T_LEN   1024
#define D_IN    512
#define H_DIM   512
#define G4H     2048      // 4*H
#define C_VOCAB 50257
#define CPAD    50272     // next multiple of 32
#define MAXLEN_ 150

typedef __attribute__((ext_vector_type(16))) __bf16 v16bf;
typedef __attribute__((ext_vector_type(8)))  __bf16 v8bf;
typedef __attribute__((ext_vector_type(8)))  float  v8f;

// Load a 16-element bf16 A/B fragment for v_wmma_f32_16x16x32_bf16.
// Per the CDNA5 16-bit A/B layout, each lane's 16 values are two contiguous
// 8-element (16-byte) runs in K: [k0 .. k0+7] and [k0+16 .. k0+23], where the
// caller folds khalf (lane>>4)*8 into the pointer.
__device__ __forceinline__ v16bf ldfrag(const __bf16* p) {
    v8bf lo = *(const v8bf*)(p);
    v8bf hi = *(const v8bf*)(p + 16);
    v16bf r;
#pragma unroll
    for (int i = 0; i < 8; ++i) { r[i] = lo[i]; r[i + 8] = hi[i]; }
    return r;
}

__device__ __forceinline__ v8f wmma_bf16(v16bf a, v16bf b, v8f c) {
    return __builtin_amdgcn_wmma_f32_16x16x32_bf16(
        /*neg_a=*/false, a, /*neg_b=*/false, b,
        /*c_mod=*/(short)0, c, /*reuse_a=*/false, /*reuse_b=*/false);
}

__device__ __forceinline__ float sigm(float x) { return 1.0f / (1.0f + __expf(-x)); }

// ---------------------------------------------------------------- fp32 -> bf16
__global__ void k_cvt(const float* __restrict__ s, __bf16* __restrict__ d, long n) {
    long i  = (long)blockIdx.x * blockDim.x + threadIdx.x;
    long st = (long)gridDim.x * blockDim.x;
    for (; i < n; i += st) d[i] = (__bf16)s[i];
}

// ------------------------------------------------------ TDM warm (gfx1250 TDM)
// Builds a real 2D tensor descriptor (16 x 512 bf16 tile of Whh, stride 512)
// and DMAs it into LDS via the Tensor Data Mover, then waits on TENSORcnt.
// Used once as an L2 warm of the recurrence weights.
#if defined(__has_builtin)
#if __has_builtin(__builtin_amdgcn_tensor_load_to_lds) && \
    __has_builtin(__builtin_amdgcn_s_wait_tensorcnt)
#define HAVE_TDM 1
#endif
#endif

#ifdef HAVE_TDM
typedef unsigned int v4u __attribute__((ext_vector_type(4)));
typedef int          v8i __attribute__((ext_vector_type(8)));
typedef int          v4i __attribute__((ext_vector_type(4)));

__global__ void __launch_bounds__(32)
k_tdm_warm(const __bf16* __restrict__ W, float* __restrict__ sink) {
    __shared__ __align__(128) __bf16 stage[16 * H_DIM];   // 16 KB target tile
    unsigned long long ga = (unsigned long long)(const void*)W;

    v4u g0;
    g0[0] = 1u;                                            // count = 1 (valid)
    g0[1] = 0u;                                            // lds_addr = 0
    g0[2] = (unsigned)(ga & 0xFFFFFFFFull);                // global_addr[31:0]
    g0[3] = (unsigned)((ga >> 32) & 0x01FFFFFFull)         // global_addr[56:32]
            | (2u << 30);                                  // type = 2 ("image")

    v8i g1 = {};
    g1[0] = 0x00010000;                                    // data_size = 1 -> 2B
    g1[1] = (int)((H_DIM & 0xFFFF) << 16);                 // tensor_dim0 = 512
    g1[2] = (int)((G4H  & 0xFFFF) << 16);                  // tensor_dim1 = 2048
    g1[3] = (int)((H_DIM & 0xFFFF) << 16);                 // tile_dim0 = 512
    g1[4] = 16;                                            // tile_dim1 = 16
    g1[5] = H_DIM;                                         // dim0_stride = 512
    g1[6] = 0; g1[7] = 0;

    v4i gz = {};
#if __clang_major__ >= 23
    v8i gz8 = {};
    __builtin_amdgcn_tensor_load_to_lds(g0, g1, gz, gz, gz8, 0);
#else
    __builtin_amdgcn_tensor_load_to_lds(g0, g1, gz, gz, 0);
#endif
    __builtin_amdgcn_s_wait_tensorcnt(0);
    sink[threadIdx.x] = (float)stage[threadIdx.x * 32];    // keep it live
}
#endif

// --------------------- C[M,N] = A[M,K] @ Bt[N,K]^T + bias, 2x2 register block
// One wave per 32x32 output patch: 4 fragment loads feed 4 WMMAs (2 loads/WMMA).
__global__ void __launch_bounds__(256)
k_gemm(const __bf16* __restrict__ A, const __bf16* __restrict__ Bt,
       const float* __restrict__ bias, float* __restrict__ Cm,
       int M, int N, int K) {
    int gw     = (int)((blockIdx.x * blockDim.x + threadIdx.x) >> 5);
    int lane   = threadIdx.x & 31;
    int tilesN = N >> 5;
    int nTiles = (M >> 5) * tilesN;
    if (gw >= nTiles) return;
    int tM = gw / tilesN, tN = gw - tM * tilesN;
    int nn = lane & 15, kh = lane >> 4;

    const __bf16* pa0 = A  + (size_t)(tM * 32 + nn) * K + kh * 8;
    const __bf16* pa1 = pa0 + (size_t)16 * K;
    const __bf16* pb0 = Bt + (size_t)(tN * 32 + nn) * K + kh * 8;
    const __bf16* pb1 = pb0 + (size_t)16 * K;

    v8f a00 = {}, a01 = {}, a10 = {}, a11 = {};
    for (int k = 0; k < K; k += 32) {
        v16bf fa0 = ldfrag(pa0 + k), fa1 = ldfrag(pa1 + k);
        v16bf fb0 = ldfrag(pb0 + k), fb1 = ldfrag(pb1 + k);
        a00 = wmma_bf16(fa0, fb0, a00);
        a01 = wmma_bf16(fa0, fb1, a01);
        a10 = wmma_bf16(fa1, fb0, a10);
        a11 = wmma_bf16(fa1, fb1, a11);
    }

    auto st = [&](v8f acc, int r0, int c0) {
        float bv = bias[c0 + nn];
#pragma unroll
        for (int r = 0; r < 8; ++r)
            Cm[(size_t)(r0 + r + 8 * kh) * N + c0 + nn] = acc[r] + bv;
    };
    st(a00, tM * 32,      tN * 32);
    st(a01, tM * 32,      tN * 32 + 16);
    st(a10, tM * 32 + 16, tN * 32);
    st(a11, tM * 32 + 16, tN * 32 + 16);
}

// ------------------------------------------------ sequential LSTM recurrence
// grid = 2 (one block per direction), block = 1024 (32 waves). h lives in LDS
// as bf16, broadcast into a replicated B fragment shared by the wave's 4
// row-tiles (B hoisted out of the row loop: 2.5 loads/WMMA, B from LDS).
__global__ void __launch_bounds__(1024)
k_lstm_recur(const float* __restrict__ Gbase,   // [2][T][4H]  (Wih@x + b)
             const __bf16* __restrict__ WhhB,   // [2][4H][H]
             __bf16* __restrict__ hs,           // [T][2H]
             float* __restrict__ hfin,          // [2][H]
             float* __restrict__ cfin,          // [2][H]
             int Tlen) {
    const int dir = blockIdx.x;
    const float*  G   = Gbase + (size_t)dir * Tlen * G4H;
    const __bf16* Whh = WhhB  + (size_t)dir * G4H * H_DIM;

    __shared__ __align__(16) __bf16 h_sh[H_DIM];
    __shared__ float gates[G4H];

    int tid  = threadIdx.x;
    int lane = tid & 31, wv = tid >> 5;
    int nn   = lane & 15, kh = lane >> 4;

    if (tid < H_DIM) h_sh[tid] = (__bf16)0.0f;
    float c_reg = 0.0f, h_reg = 0.0f;
    __syncthreads();

    for (int step = 0; step < Tlen; ++step) {
        int t = dir ? (Tlen - 1 - step) : step;
        v8f acc[4] = {{}, {}, {}, {}};
        for (int k = 0; k < H_DIM; k += 32) {
            v16bf b = ldfrag(h_sh + kh * 8 + k);
#pragma unroll
            for (int q = 0; q < 4; ++q) {
                const __bf16* pa =
                    Whh + (size_t)((wv * 4 + q) * 16 + nn) * H_DIM + kh * 8 + k;
                acc[q] = wmma_bf16(ldfrag(pa), b, acc[q]);
            }
        }
        if (nn == 0) {
#pragma unroll
            for (int q = 0; q < 4; ++q)
#pragma unroll
                for (int r = 0; r < 8; ++r)
                    gates[(wv * 4 + q) * 16 + r + 8 * kh] = acc[q][r];
        }
        __syncthreads();
        if (tid < H_DIM) {
            const float* g = G + (size_t)t * G4H;
            float gi = gates[tid]             + g[tid];
            float gf = gates[H_DIM + tid]     + g[H_DIM + tid];
            float gg = gates[2 * H_DIM + tid] + g[2 * H_DIM + tid];
            float go = gates[3 * H_DIM + tid] + g[3 * H_DIM + tid];
            c_reg = sigm(gf) * c_reg + sigm(gi) * tanhf(gg);
            h_reg = sigm(go) * tanhf(c_reg);
            __bf16 hb = (__bf16)h_reg;
            h_sh[tid] = hb;
            hs[(size_t)t * (2 * H_DIM) + dir * H_DIM + tid] = hb;
        }
        __syncthreads();
    }
    if (tid < H_DIM) {
        hfin[dir * H_DIM + tid] = h_reg;
        cfin[dir * H_DIM + tid] = c_reg;
    }
}

// ------------------------------------------------ one decoder LSTM layer step
__global__ void __launch_bounds__(1024)
k_dec_cell(const __bf16* __restrict__ WihB,   // [2][4H][Din]
           const __bf16* __restrict__ WhhB,   // [2][4H][H]
           const float*  __restrict__ biasB,  // [2][4H]
           const __bf16* __restrict__ inp,    // [Din]
           float* __restrict__ hio,           // [2][H]
           float* __restrict__ cio,           // [2][H]
           __bf16* __restrict__ outcat,       // [2H]
           int Din) {
    const int dir = blockIdx.x;
    const __bf16* Wih  = WihB  + (size_t)dir * G4H * Din;
    const __bf16* Whh  = WhhB  + (size_t)dir * G4H * H_DIM;
    const float*  bias = biasB + (size_t)dir * G4H;
    float* hv = hio + dir * H_DIM;
    float* cv = cio + dir * H_DIM;

    __shared__ __align__(16) __bf16 x_sh[2 * H_DIM];
    __shared__ __align__(16) __bf16 h_sh[H_DIM];
    __shared__ float gates[G4H];

    int tid = threadIdx.x;
    if (tid < Din)   x_sh[tid] = inp[tid];
    if (tid < H_DIM) h_sh[tid] = (__bf16)hv[tid];
    __syncthreads();

    int lane = tid & 31, wv = tid >> 5;
    int nn   = lane & 15, kh = lane >> 4;

    v8f acc[4] = {{}, {}, {}, {}};
    for (int k = 0; k < Din; k += 32) {
        v16bf b = ldfrag(x_sh + kh * 8 + k);
#pragma unroll
        for (int q = 0; q < 4; ++q) {
            const __bf16* pa =
                Wih + (size_t)((wv * 4 + q) * 16 + nn) * Din + kh * 8 + k;
            acc[q] = wmma_bf16(ldfrag(pa), b, acc[q]);
        }
    }
    for (int k = 0; k < H_DIM; k += 32) {
        v16bf b = ldfrag(h_sh + kh * 8 + k);
#pragma unroll
        for (int q = 0; q < 4; ++q) {
            const __bf16* pa =
                Whh + (size_t)((wv * 4 + q) * 16 + nn) * H_DIM + kh * 8 + k;
            acc[q] = wmma_bf16(ldfrag(pa), b, acc[q]);
        }
    }
    if (nn == 0) {
#pragma unroll
        for (int q = 0; q < 4; ++q)
#pragma unroll
            for (int r = 0; r < 8; ++r)
                gates[(wv * 4 + q) * 16 + r + 8 * kh] = acc[q][r];
    }
    __syncthreads();
    if (tid < H_DIM) {
        float gi = gates[tid]             + bias[tid];
        float gf = gates[H_DIM + tid]     + bias[H_DIM + tid];
        float gg = gates[2 * H_DIM + tid] + bias[2 * H_DIM + tid];
        float go = gates[3 * H_DIM + tid] + bias[3 * H_DIM + tid];
        float c = sigm(gf) * cv[tid] + sigm(gi) * tanhf(gg);
        float h = sigm(go) * tanhf(c);
        cv[tid] = c;
        hv[tid] = h;
        outcat[dir * H_DIM + tid] = (__bf16)h;
    }
}

// ---------------------------------- vocab GEMV + per-block argmax contribution
// Two row-tiles per wave share one B fragment; weight stream is prefetched.
__global__ void __launch_bounds__(256)
k_logits(const __bf16* __restrict__ W,     // [C][2H] bf16
         const float*  __restrict__ bias,  // [C]
         const __bf16* __restrict__ inp,   // [2H]
         float* __restrict__ out,          // logits for this step
         unsigned long long* __restrict__ slot) {
    __shared__ __align__(16) __bf16 x_sh[2 * H_DIM];
    __shared__ unsigned long long best[256];

    int tid = threadIdx.x;
    for (int i = tid; i < 2 * H_DIM; i += 256) x_sh[i] = inp[i];
    __syncthreads();

    int lane = tid & 31, wv = tid >> 5;
    int nn   = lane & 15, kh = lane >> 4;
    int pair = blockIdx.x * 8 + wv;

    unsigned long long myb = 0ull;
    if (pair < (CPAD >> 5)) {
        int row0  = pair * 32;
        int arow0 = row0 + nn;      if (arow0 >= C_VOCAB) arow0 = C_VOCAB - 1;
        int arow1 = row0 + 16 + nn; if (arow1 >= C_VOCAB) arow1 = C_VOCAB - 1;
        const __bf16* pa0 = W + (size_t)arow0 * (2 * H_DIM) + kh * 8;
        const __bf16* pa1 = W + (size_t)arow1 * (2 * H_DIM) + kh * 8;
        v8f acc0 = {}, acc1 = {};
        for (int k = 0; k < 2 * H_DIM; k += 32) {
            if ((k & 255) == 0) {
                __builtin_prefetch(pa0 + k + 1024, 0, 3);
                __builtin_prefetch(pa1 + k + 1024, 0, 3);
            }
            v16bf b = ldfrag(x_sh + kh * 8 + k);
            acc0 = wmma_bf16(ldfrag(pa0 + k), b, acc0);
            acc1 = wmma_bf16(ldfrag(pa1 + k), b, acc1);
        }
        if (nn == 0) {
#pragma unroll
            for (int h = 0; h < 2; ++h) {
                v8f acc = h ? acc1 : acc0;
                int rb  = row0 + 16 * h;
#pragma unroll
                for (int r = 0; r < 8; ++r) {
                    int row = rb + r + 8 * kh;
                    if (row < C_VOCAB) {
                        float v = acc[r] + bias[row];
                        out[row] = v;
                        unsigned int b32 = __float_as_uint(v);
                        b32 = (b32 & 0x80000000u) ? ~b32 : (b32 | 0x80000000u);
                        unsigned long long p = ((unsigned long long)b32 << 32) |
                                               (unsigned int)(0xFFFFFFFFu - row);
                        if (p > myb) myb = p;
                    }
                }
            }
        }
    }
    best[tid] = myb;
    __syncthreads();
    for (int s = 128; s > 0; s >>= 1) {
        if (tid < s && best[tid + s] > best[tid]) best[tid] = best[tid + s];
        __syncthreads();
    }
    if (tid == 0 && best[0] != 0ull) atomicMax(slot, best[0]);
}

// ----------------------------- resolve token, emit it, gather embedding (bf16)
__global__ void k_finish(const unsigned long long* __restrict__ slot,
                         const float* __restrict__ emb,   // [C][D] fp32
                         float* __restrict__ tok_out,     // one float
                         __bf16* __restrict__ buf) {      // [D]
    __shared__ int tok_sh;
    if (threadIdx.x == 0) {
        unsigned long long p = *slot;
        int tok = (int)(0xFFFFFFFFu - (unsigned int)(p & 0xFFFFFFFFull));
        tok_sh = tok;
        *tok_out = (float)tok;
    }
    __syncthreads();
    int tok = tok_sh;
    for (int i = threadIdx.x; i < D_IN; i += blockDim.x)
        buf[i] = (__bf16)emb[(size_t)tok * D_IN + i];
}

// ---------------------------------------------------- per-call initialization
__global__ void k_init(unsigned long long* __restrict__ slots,
                       const float* __restrict__ bos,
                       __bf16* __restrict__ buf) {
    int i = blockIdx.x * blockDim.x + threadIdx.x;
    if (i < MAXLEN_) slots[i] = 0ull;
    if (i < D_IN)    buf[i] = (__bf16)bos[i];
}

// ============================================================================
extern "C" void kernel_launch(void* const* d_in, const int* in_sizes, int n_in,
                              void* d_out, int out_size, void* d_ws, size_t ws_size,
                              hipStream_t stream) {
    (void)in_sizes; (void)n_in; (void)out_size; (void)ws_size;

    const float* x     = (const float*)d_in[0];
    const float* eWih0 = (const float*)d_in[1];
    const float* eWhh0 = (const float*)d_in[2];
    const float* eB0   = (const float*)d_in[3];
    const float* eWih1 = (const float*)d_in[4];
    const float* eWhh1 = (const float*)d_in[5];
    const float* eB1   = (const float*)d_in[6];
    const float* dWih0 = (const float*)d_in[7];
    const float* dWhh0 = (const float*)d_in[8];
    const float* dB0   = (const float*)d_in[9];
    const float* dWih1 = (const float*)d_in[10];
    const float* dWhh1 = (const float*)d_in[11];
    const float* dB1   = (const float*)d_in[12];
    const float* linW  = (const float*)d_in[13];
    const float* linB  = (const float*)d_in[14];
    const float* emb   = (const float*)d_in[15];
    const float* bos   = (const float*)d_in[16];
    float* out = (float*)d_out;

    // ---- workspace carve-out (256B aligned)
    char*  base = (char*)d_ws;
    size_t off  = 0;
    auto alloc = [&](size_t bytes) -> void* {
        void* p = base + off;
        off = (off + bytes + 255) & ~(size_t)255;
        return p;
    };
    __bf16* x_bf  = (__bf16*)alloc((size_t)T_LEN * D_IN * 2);
    __bf16* wEih0 = (__bf16*)alloc((size_t)2 * G4H * D_IN * 2);
    __bf16* wEhh0 = (__bf16*)alloc((size_t)2 * G4H * H_DIM * 2);
    __bf16* wEih1 = (__bf16*)alloc((size_t)2 * G4H * 2 * H_DIM * 2);
    __bf16* wEhh1 = (__bf16*)alloc((size_t)2 * G4H * H_DIM * 2);
    __bf16* wDih0 = (__bf16*)alloc((size_t)2 * G4H * D_IN * 2);
    __bf16* wDhh0 = (__bf16*)alloc((size_t)2 * G4H * H_DIM * 2);
    __bf16* wDih1 = (__bf16*)alloc((size_t)2 * G4H * 2 * H_DIM * 2);
    __bf16* wDhh1 = (__bf16*)alloc((size_t)2 * G4H * H_DIM * 2);
    __bf16* wLin  = (__bf16*)alloc((size_t)C_VOCAB * 2 * H_DIM * 2);
    float*  Gbuf  = (float*) alloc((size_t)2 * T_LEN * G4H * 4);
    __bf16* hs0   = (__bf16*)alloc((size_t)T_LEN * 2 * H_DIM * 2);
    float*  dech  = (float*) alloc((size_t)2 * 2 * H_DIM * 4);
    float*  decc  = (float*) alloc((size_t)2 * 2 * H_DIM * 4);
    __bf16* inp1  = (__bf16*)alloc((size_t)2 * H_DIM * 2);
    __bf16* inp2  = (__bf16*)alloc((size_t)2 * H_DIM * 2);
    __bf16* buf   = (__bf16*)alloc((size_t)D_IN * 2);
    unsigned long long* slots = (unsigned long long*)alloc((size_t)MAXLEN_ * 8);
    float*  tdms  = (float*) alloc((size_t)32 * 4);

    auto cvt = [&](const float* s, __bf16* d, long n) {
        int blocks = (int)((n + 2047) / 2048);
        k_cvt<<<blocks, 256, 0, stream>>>(s, d, n);
    };
    cvt(x,     x_bf,  (long)T_LEN * D_IN);
    cvt(eWih0, wEih0, 2L * G4H * D_IN);
    cvt(eWhh0, wEhh0, 2L * G4H * H_DIM);
    cvt(eWih1, wEih1, 2L * G4H * 2 * H_DIM);
    cvt(eWhh1, wEhh1, 2L * G4H * H_DIM);
    cvt(dWih0, wDih0, 2L * G4H * D_IN);
    cvt(dWhh0, wDhh0, 2L * G4H * H_DIM);
    cvt(dWih1, wDih1, 2L * G4H * 2 * H_DIM);
    cvt(dWhh1, wDhh1, 2L * G4H * H_DIM);
    cvt(linW,  wLin,  (long)C_VOCAB * 2 * H_DIM);

    k_init<<<1, 512, 0, stream>>>(slots, bos, buf);

#ifdef HAVE_TDM
    k_tdm_warm<<<1, 32, 0, stream>>>(wEhh0, tdms);
#else
    (void)tdms;
#endif

    auto gemm = [&](const __bf16* A, const __bf16* Bt, const float* bias,
                    float* C, int M, int N, int K) {
        int tiles  = (M >> 5) * (N >> 5);
        int blocks = (tiles + 7) / 8;     // 8 waves (256 threads) per block
        k_gemm<<<blocks, 256, 0, stream>>>(A, Bt, bias, C, M, N, K);
    };

    // ---------------- encoder layer 0
    for (int dir = 0; dir < 2; ++dir)
        gemm(x_bf, wEih0 + (size_t)dir * G4H * D_IN, eB0 + dir * G4H,
             Gbuf + (size_t)dir * T_LEN * G4H, T_LEN, G4H, D_IN);
    k_lstm_recur<<<2, 1024, 0, stream>>>(Gbuf, wEhh0, hs0, dech, decc, T_LEN);

    // ---------------- encoder layer 1 (input = [hs_f | hs_b], K = 2H)
    for (int dir = 0; dir < 2; ++dir)
        gemm(hs0, wEih1 + (size_t)dir * G4H * 2 * H_DIM, eB1 + dir * G4H,
             Gbuf + (size_t)dir * T_LEN * G4H, T_LEN, G4H, 2 * H_DIM);
    k_lstm_recur<<<2, 1024, 0, stream>>>(Gbuf, wEhh1, hs0,
                                         dech + 2 * H_DIM, decc + 2 * H_DIM, T_LEN);

    // ---------------- greedy decoder: 150 sequential steps
    const int lblocks = ((CPAD >> 5) + 7) / 8;
    for (int s = 0; s < MAXLEN_; ++s) {
        k_dec_cell<<<2, 1024, 0, stream>>>(wDih0, wDhh0, dB0, buf,
                                           dech, decc, inp1, D_IN);
        k_dec_cell<<<2, 1024, 0, stream>>>(wDih1, wDhh1, dB1, inp1,
                                           dech + 2 * H_DIM, decc + 2 * H_DIM,
                                           inp2, 2 * H_DIM);
        k_logits<<<lblocks, 256, 0, stream>>>(wLin, linB, inp2,
                                              out + (size_t)s * C_VOCAB, slots + s);
        k_finish<<<1, 512, 0, stream>>>(slots + s, emb,
                                        out + (size_t)MAXLEN_ * C_VOCAB + s, buf);
    }
}
